// SelfAttention_58617713656547
// MI455X (gfx1250) — matrix operand order repaired and checked by reading, not compiled, not run on hardware
//
#include <hip/hip_runtime.h>

typedef __attribute__((ext_vector_type(16))) __bf16 v16bf;
typedef __attribute__((ext_vector_type(8)))  float  v8f;
typedef __attribute__((ext_vector_type(8)))  __bf16 bf16x8;
typedef __attribute__((ext_vector_type(2)))  __bf16 bf16x2;

union ABfrag {
  v16bf  v;
  bf16x8 h[2];
  bf16x2 p[8];
  __bf16 e[16];
};

constexpr int   BATCH  = 2;
constexpr int   HEADS  = 16;
constexpr int   SEQ    = 2048;
constexpr int   DHEAD  = 64;
constexpr int   QT     = 16;                 // query rows per block
constexpr float SCALE  = 1.0f / 1024.0f;     // reference divides by d_model
constexpr float NEG_INF_F = -1.0e9f;

constexpr size_t NQKV  = (size_t)BATCH * HEADS * SEQ * DHEAD;   // 4,194,304

constexpr int SC_ELEMS   = QT * SEQ;         // f32 score buffer elems
// scores(f32) + P(bf16) + red[128] + rowm[16] + rowinv[16]
constexpr int SMEM_BYTES = SC_ELEMS * 4 + SC_ELEMS * 2
                         + 128 * 4 + 16 * 4 + 16 * 4;   // = 197,248 B < 320 KB

// ---------------- preprocessing: bf16 + WMMA-friendly layouts ----------------

// Q f32 [bh][s][d]  ->  Qbf bf16 [bh][s][d]  (same layout)
__global__ __launch_bounds__(256)
void cvt_q_kernel(const float* __restrict__ Q, __bf16* __restrict__ Qbf) {
  const size_t i = (size_t)blockIdx.x * 256 + threadIdx.x;
  if (i < NQKV) Qbf[i] = (__bf16)Q[i];
}

// K f32 [bh][s][d] -> Kp bf16 pair-interleaved over d: [bh][d/2][s][2]
// One thread per output pair; out offset == 2*idx by construction.
__global__ __launch_bounds__(256)
void cvt_kp_kernel(const float* __restrict__ K, __bf16* __restrict__ Kp) {
  const size_t idx = (size_t)blockIdx.x * 256 + threadIdx.x;
  if (idx >= NQKV / 2) return;
  const int s  = (int)(idx & (SEQ - 1));
  const size_t t = idx >> 11;               // SEQ == 2048
  const int dp = (int)(t & 31);             // DHEAD/2 == 32
  const size_t bh = t >> 5;
  const float2 f = *(const float2*)&K[(bh * SEQ + s) * DHEAD + 2 * dp];
  bf16x2 o; o.x = (__bf16)f.x; o.y = (__bf16)f.y;
  *(bf16x2*)&Kp[idx * 2] = o;
}

// V f32 [bh][s][d] -> Vp bf16 pair-interleaved over s: [bh][s/2][d][2]
__global__ __launch_bounds__(256)
void cvt_vp_kernel(const float* __restrict__ V, __bf16* __restrict__ Vp) {
  const size_t idx = (size_t)blockIdx.x * 256 + threadIdx.x;
  if (idx >= NQKV / 2) return;
  const int d  = (int)(idx & (DHEAD - 1));
  const size_t t = idx >> 6;
  const int sp = (int)(t & 1023);           // SEQ/2 == 1024
  const size_t bh = t >> 10;
  const size_t in0 = (bh * SEQ + 2 * sp) * DHEAD + d;
  bf16x2 o; o.x = (__bf16)V[in0]; o.y = (__bf16)V[in0 + DHEAD];
  *(bf16x2*)&Vp[idx * 2] = o;
}

// ---------------------------- fused attention --------------------------------

__global__ __launch_bounds__(128)
void attn_fused_bf16_wmma(const __bf16* __restrict__ Qbf,
                          const __bf16* __restrict__ Kp,
                          const __bf16* __restrict__ Vp,
                          const int*    __restrict__ M,
                          float* __restrict__ Out,
                          float* __restrict__ Attn)
{
  extern __shared__ char smem[];
  float*  sc     = (float*)smem;                       // [16][2048] f32 scores
  __bf16* pbf    = (__bf16*)(smem + SC_ELEMS * 4);     // [16][2048] bf16 probs
  float*  red    = (float*)(pbf + SC_ELEMS);           // [128] reduction scratch
  float*  rowm   = red + 128;                          // [16] row max
  float*  rowinv = rowm + 16;                          // [16] 1/row_sum

  const int tid  = threadIdx.x;          // 0..127 (4 waves of wave32)
  const int wave = tid >> 5;
  const int lane = tid & 31;
  const int hi   = lane >> 4;            // half-wave selector (WMMA layout)
  const int nl   = lane & 15;            // M (A) or N (B/C) index

  const int blk = blockIdx.x;
  const int bh  = blk >> 7;              // / (SEQ/QT) = 128
  const int q0  = (blk & 127) * QT;
  const int b   = bh >> 4;               // / HEADS

  const size_t qkv_base = (size_t)bh * SEQ * DHEAD;

  // ---- hoisted A fragments of Q for d in [0,32) and [32,64) ----
  // A 16x32 bf16 in this code's contiguous k order (B matches): lanes 0-15 hold K 0-15, lanes 16-31 hold K 16-31
  ABfrag a0, a1;
  {
    const __bf16* qr = &Qbf[qkv_base + (size_t)(q0 + nl) * DHEAD];
    a0.h[0] = *(const bf16x8*)&qr[ 0 + hi * 16];
    a0.h[1] = *(const bf16x8*)&qr[ 8 + hi * 16];
    a1.h[0] = *(const bf16x8*)&qr[32 + hi * 16];
    a1.h[1] = *(const bf16x8*)&qr[40 + hi * 16];
  }

  // ---- Phase 1: scores = (Q K^T) * SCALE, masked, into LDS f32 ----
  // Kp layout [bh][dp=32][s][2]: B VGPR j for fragment db is one b32 load.
  const __bf16* kp0 = &Kp[(((size_t)bh * 32 +      hi * 8) * SEQ) * 2];
  const __bf16* kp1 = &Kp[(((size_t)bh * 32 + 16 + hi * 8) * SEQ) * 2];
  const int*    mrow_base = &M[(size_t)b * SEQ * SEQ + (size_t)(q0 + hi * 8) * SEQ];

  for (int kt = wave; kt < SEQ / 16; kt += 4) {
    const int k0  = kt * 16;
    const int col = k0 + nl;
    ABfrag b0, b1;
#pragma unroll
    for (int j = 0; j < 8; ++j) {                    // B VGPR j: K = 2j, 2j+1
      b0.p[j] = *(const bf16x2*)&kp0[((size_t)j * SEQ + col) * 2];
      b1.p[j] = *(const bf16x2*)&kp1[((size_t)j * SEQ + col) * 2];
    }
    v8f c = {};
    c = __builtin_amdgcn_wmma_f32_16x16x32_bf16(false, a0.v, false, b0.v,
                                                (short)0, c, false, false);
    c = __builtin_amdgcn_wmma_f32_16x16x32_bf16(false, a1.v, false, b1.v,
                                                (short)0, c, false, false);
#pragma unroll
    for (int r = 0; r < 8; ++r) {                    // C VGPR r: M = r + hi*8
      const int mrow = r + hi * 8;
      const int mv   = mrow_base[(size_t)r * SEQ + col];
      sc[mrow * SEQ + col] = (mv == 0) ? NEG_INF_F : c[r] * SCALE;
    }
  }
  __syncthreads();

  // ---- Phase 2a: row max (8 threads per row) ----
  {
    const int r = tid >> 3, sub = tid & 7;
    float m = -3.4e38f;
    for (int cc = sub; cc < SEQ; cc += 8) m = fmaxf(m, sc[r * SEQ + cc]);
    red[tid] = m;
  }
  __syncthreads();
  if ((tid & 7) == 0) {
    float m = red[tid];
#pragma unroll
    for (int j = 1; j < 8; ++j) m = fmaxf(m, red[tid + j]);
    rowm[tid >> 3] = m;
  }
  __syncthreads();

  // ---- Phase 2b: row sum of exp ----
  {
    const int r = tid >> 3, sub = tid & 7;
    const float m = rowm[r];
    float s = 0.f;
    for (int cc = sub; cc < SEQ; cc += 8) s += __expf(sc[r * SEQ + cc] - m);
    red[tid] = s;
  }
  __syncthreads();
  if ((tid & 7) == 0) {
    float s = red[tid];
#pragma unroll
    for (int j = 1; j < 8; ++j) s += red[tid + j];
    rowinv[tid >> 3] = 1.0f / s;
  }
  __syncthreads();

  // ---- Phase 2c: normalize; nontemporal stream of attn (536MB, never re-read
  //      -> keep it out of L2); stage P as bf16 for the PV matmul ----
  {
    float* attn_base = &Attn[(size_t)bh * SEQ * SEQ + (size_t)q0 * SEQ];
    for (int idx = tid; idx < QT * SEQ; idx += 128) {
      const int   r = idx >> 11;                     // SEQ == 2048
      const float p = __expf(sc[idx] - rowm[r]) * rowinv[r];
      __builtin_nontemporal_store(p, attn_base + idx);  // rows contiguous
      pbf[idx] = (__bf16)p;
    }
  }
  __syncthreads();

  // ---- Phase 3: out = P @ V ; each wave owns a 16-wide d-chunk ----
  // Vp layout [bh][sp=1024][d][2]: B VGPR j for K-rows (kb+hi*16+2j, +1).
  const int d0 = wave * 16;
  const __bf16* vp = &Vp[(((size_t)bh * (SEQ / 2)) * DHEAD + d0 + nl) * 2];
  v8f acc = {};
  for (int kt = 0; kt < SEQ / 32; ++kt) {
    const int kb = kt * 32;
    ABfrag a;
    const __bf16* pr = &pbf[nl * SEQ + kb];
    a.h[0] = *(const bf16x8*)&pr[ 0 + hi * 16];
    a.h[1] = *(const bf16x8*)&pr[ 8 + hi * 16];
    ABfrag bb;
    const int spbase = kb / 2 + hi * 8;              // seq-pair index
#pragma unroll
    for (int j = 0; j < 8; ++j)
      bb.p[j] = *(const bf16x2*)&vp[(size_t)(spbase + j) * DHEAD * 2];
    acc = __builtin_amdgcn_wmma_f32_16x16x32_bf16(false, a.v, false, bb.v,
                                                  (short)0, acc, false, false);
  }
#pragma unroll
  for (int r = 0; r < 8; ++r) {
    const int orow = q0 + r + hi * 8;
    Out[qkv_base + (size_t)orow * DHEAD + d0 + nl] = acc[r];
  }
}

// ------------------------------- launcher ------------------------------------

extern "C" void kernel_launch(void* const* d_in, const int* in_sizes, int n_in,
                              void* d_out, int out_size, void* d_ws, size_t ws_size,
                              hipStream_t stream) {
  const float* q    = (const float*)d_in[0];
  const float* k    = (const float*)d_in[1];
  const float* v    = (const float*)d_in[2];
  const int*   mask = (const int*)d_in[3];

  float* out  = (float*)d_out;                                   // [B,H,S,D]
  float* attn = out + NQKV;                                      // [B,H,S,S]

  __bf16* qbf = (__bf16*)d_ws;                                   // 8 MB
  __bf16* kp  = qbf + NQKV;                                      // 8 MB
  __bf16* vp  = kp  + NQKV;                                      // 8 MB

  // Preprocess: bf16 conversion + WMMA-fragment-friendly pair-interleaving.
  {
    const int thr = 256;
    const int blocksQ = (int)((NQKV + thr - 1) / thr);
    const int blocksP = (int)((NQKV / 2 + thr - 1) / thr);
    cvt_q_kernel <<<blocksQ, thr, 0, stream>>>(q, qbf);
    cvt_kp_kernel<<<blocksP, thr, 0, stream>>>(k, kp);
    cvt_vp_kernel<<<blocksP, thr, 0, stream>>>(v, vp);
  }

  // Allow >64KB dynamic LDS (WGP supports up to 320KB per workgroup).
  (void)hipFuncSetAttribute((const void*)attn_fused_bf16_wmma,
                            hipFuncAttributeMaxDynamicSharedMemorySize,
                            SMEM_BYTES);

  const int blocks = BATCH * HEADS * (SEQ / QT);                 // 4096
  attn_fused_bf16_wmma<<<blocks, 128, SMEM_BYTES, stream>>>(qbf, kp, vp, mask,
                                                            out, attn);
}
